// NativeMambaBlock_42502996361233
// MI455X (gfx1250) — compile-verified, matrix operand order checked by reading, not run on hardware
//
#include <hip/hip_runtime.h>
#include <hip/hip_bf16.h>
#include <math.h>
#include <stdint.h>

typedef __bf16 bf16;
typedef __attribute__((ext_vector_type(16))) __bf16 v16bf;
typedef __attribute__((ext_vector_type(8)))  float  v8f;
typedef int v4i __attribute__((vector_size(16)));

#define D_MODEL 1024
#define D_INNER 2048
#define NSTATE  16
#define KCONV   4
#define RANK    64
#define BATCH   2
#define LSEQ    2048
#define MROWS   (BATCH*LSEQ)   /* 4096 */
#define DBL_PAD 128            /* RANK + 2*N = 96 padded to 128 */

/* ---- CDNA5 async global->LDS path (ASYNCcnt-tracked DMA-style copy) ---- */
#if __has_builtin(__builtin_amdgcn_global_load_async_to_lds_b128)
#define HAS_ASYNC 1
#else
#define HAS_ASYNC 0
#endif

#if HAS_ASYNC
#define GLD_ASYNC_B128(gp, lp) \
    __builtin_amdgcn_global_load_async_to_lds_b128((v4i*)(gp), (v4i*)(lp), 0, 0)
#endif

#if __has_builtin(__builtin_amdgcn_s_wait_asynccnt)
#define WAIT_ASYNC(n) __builtin_amdgcn_s_wait_asynccnt(n)
#else
#define WAIT_ASYNC(n) asm volatile("s_wait_asynccnt %0" :: "i"(n) : "memory")
#endif

/* ------------------------- WMMA GEMM ------------------------------ */
/* C[m,n] = sum_k A[m,k]*B[n,k]  (A: M x K bf16, B: N x K bf16 row-major,
 * exactly how the weights are stored). Workgroup tile 128x128, K-step 32,
 * 8 waves -> each wave 32x64 via 2x4 v_wmma_f32_16x16x32_bf16.
 * Double-buffered LDS fed by global_load_async_to_lds_b128. */
#define TM 128
#define TN 128
#define TKK 32
#define LDSS 40   /* LDS row stride in bf16 elements (pad vs 32) */

__global__ __launch_bounds__(256)
void gemm_wmma(const bf16* __restrict__ A, int lda,
               const bf16* __restrict__ B, int ldb,
               float* Cf, bf16* Cb, int ldc, int coloff, int K,
               const float* __restrict__ bias,
               const float* __restrict__ resid)
{
    __shared__ __align__(16) bf16 lsA[2][TM][LDSS];
    __shared__ __align__(16) bf16 lsB[2][TN][LDSS];

    const int tid   = threadIdx.x;
    const int lane  = tid & 31;
    const int wid   = tid >> 5;
    const int wm    = wid & 3;     /* 0..3 -> 32-row slab  */
    const int wn    = wid >> 2;    /* 0..1 -> 64-col slab  */
    const int lrow  = lane & 15;
    const int lhalf = lane >> 4;
    const int kb0   = lhalf * 8;   /* K base per ISA 16-bit A/B layout */

    const int blockM = blockIdx.y * TM;
    const int blockN = blockIdx.x * TN;

    /* tile staging: 2 threads per tile row, 16 bf16 (= 2 x b128) each */
    const int gr = tid >> 1;
    const int gk = (tid & 1) * 16;
    const bf16* gA = A + (size_t)(blockM + gr) * lda + gk;
    const bf16* gB = B + (size_t)(blockN + gr) * ldb + gk;

    v8f acc[2][4];
#pragma unroll
    for (int i = 0; i < 2; ++i)
#pragma unroll
        for (int j = 0; j < 4; ++j)
#pragma unroll
            for (int e = 0; e < 8; ++e) acc[i][j][e] = 0.0f;

    const int nk = K / TKK;

    auto issue_tile = [&](int kt, int buf) {
#if HAS_ASYNC
        /* 4 async instructions per wave per tile (ASYNCcnt += 4) */
        GLD_ASYNC_B128(gA + (size_t)kt * TKK,     &lsA[buf][gr][gk]);
        GLD_ASYNC_B128(gA + (size_t)kt * TKK + 8, &lsA[buf][gr][gk + 8]);
        GLD_ASYNC_B128(gB + (size_t)kt * TKK,     &lsB[buf][gr][gk]);
        GLD_ASYNC_B128(gB + (size_t)kt * TKK + 8, &lsB[buf][gr][gk + 8]);
#else
        const uint4* pa = (const uint4*)(gA + (size_t)kt * TKK);
        const uint4* pb = (const uint4*)(gB + (size_t)kt * TKK);
        uint4 a0 = pa[0], a1 = pa[1], b0 = pb[0], b1 = pb[1];
        *(uint4*)&lsA[buf][gr][gk]     = a0;
        *(uint4*)&lsA[buf][gr][gk + 8] = a1;
        *(uint4*)&lsB[buf][gr][gk]     = b0;
        *(uint4*)&lsB[buf][gr][gk + 8] = b1;
#endif
    };

    issue_tile(0, 0);

    for (int kt = 0; kt < nk; ++kt) {
        const int cur = kt & 1;
        const bool pre = (kt + 1 < nk);
        if (pre) {
            issue_tile(kt + 1, cur ^ 1);
            if (kt + 2 < nk) {  /* speculative L2 prefetch of tile kt+2 */
                __builtin_prefetch(gA + (size_t)(kt + 2) * TKK, 0, 1);
                __builtin_prefetch(gB + (size_t)(kt + 2) * TKK, 0, 1);
            }
        }
#if HAS_ASYNC
        if (pre) WAIT_ASYNC(4);   /* tile kt landed; kt+1 still in flight */
        else     WAIT_ASYNC(0);
#endif
        __syncthreads();          /* all waves' tile kt visible in LDS   */

        union Frag { uint4 u[2]; v16bf v; };
        Frag fa[2], fb[4];
#pragma unroll
        for (int fm = 0; fm < 2; ++fm) {
            const bf16* p = &lsA[cur][wm * 32 + fm * 16 + lrow][kb0];
            fa[fm].u[0] = *(const uint4*)p;
            fa[fm].u[1] = *(const uint4*)(p + 16);
        }
#pragma unroll
        for (int fn = 0; fn < 4; ++fn) {
            const bf16* p = &lsB[cur][wn * 64 + fn * 16 + lrow][kb0];
            fb[fn].u[0] = *(const uint4*)p;
            fb[fn].u[1] = *(const uint4*)(p + 16);
        }
#pragma unroll
        for (int fm = 0; fm < 2; ++fm)
#pragma unroll
            for (int fn = 0; fn < 4; ++fn)
                acc[fm][fn] = __builtin_amdgcn_wmma_f32_16x16x32_bf16(
                    false, fa[fm].v, false, fb[fn].v,
                    (short)0, acc[fm][fn], false, false);

        __syncthreads();          /* done reading buf cur before it is   */
    }                             /* overwritten in the next iteration    */

    /* Epilogue. C/D layout: VGPR e holds (M = e + 8*lhalf, N = lrow). */
#pragma unroll
    for (int fm = 0; fm < 2; ++fm)
#pragma unroll
        for (int fn = 0; fn < 4; ++fn) {
            const int col = blockN + wn * 64 + fn * 16 + lrow;
#pragma unroll
            for (int e = 0; e < 8; ++e) {
                const int row = blockM + wm * 32 + fm * 16 + lhalf * 8 + e;
                float v = acc[fm][fn][e];
                const size_t off = (size_t)row * ldc + coloff + col;
                if (Cb) {
                    Cb[off] = (bf16)v;
                } else {
                    if (bias)  v += bias[col];
                    if (resid) v += resid[(size_t)row * ldc + col];
                    Cf[off] = v;
                }
            }
        }
}

/* ------------------------- elementwise kernels -------------------- */
__global__ void k_cvt(const float* __restrict__ s, bf16* __restrict__ d, long n)
{
    long i = (long)blockIdx.x * blockDim.x + threadIdx.x;
    if (i < n) d[i] = (bf16)s[i];
}

/* convert rows [0,rows_src) and zero-pad rows [rows_src,rows_dst) */
__global__ void k_cvt_pad(const float* __restrict__ s, bf16* __restrict__ d,
                          int rows_src, int cols, long total)
{
    long i = (long)blockIdx.x * blockDim.x + threadIdx.x;
    if (i >= total) return;
    long r = i / cols, c = i % cols;
    d[i] = (bf16)(r < rows_src ? s[r * (long)cols + c] : 0.0f);
}

/* strided slice fp32 -> bf16 (extract dt columns from padded dbl) */
__global__ void k_slice_bf16(const float* __restrict__ s, int lds_,
                             bf16* __restrict__ d, int ldd, int cols, long rows)
{
    long i = (long)blockIdx.x * blockDim.x + threadIdx.x;
    if (i >= rows * cols) return;
    long r = i / cols, c = i % cols;
    d[r * ldd + c] = (bf16)s[r * (long)lds_ + c];
}

/* LayerNorm over D_MODEL, emit bf16 operand for the first GEMM */
__global__ __launch_bounds__(256)
void k_ln(const float* __restrict__ x, const float* __restrict__ g,
          const float* __restrict__ b, bf16* __restrict__ xn)
{
    __shared__ float red[256];
    const int row = blockIdx.x;
    const float* xr = x + (size_t)row * D_MODEL;

    float s = 0.f;
    for (int c = threadIdx.x; c < D_MODEL; c += 256) s += xr[c];
    red[threadIdx.x] = s; __syncthreads();
    for (int o = 128; o > 0; o >>= 1) {
        if (threadIdx.x < o) red[threadIdx.x] += red[threadIdx.x + o];
        __syncthreads();
    }
    const float mu = red[0] * (1.0f / D_MODEL);
    __syncthreads();

    float s2 = 0.f;
    for (int c = threadIdx.x; c < D_MODEL; c += 256) {
        float v = xr[c] - mu; s2 += v * v;
    }
    red[threadIdx.x] = s2; __syncthreads();
    for (int o = 128; o > 0; o >>= 1) {
        if (threadIdx.x < o) red[threadIdx.x] += red[threadIdx.x + o];
        __syncthreads();
    }
    const float inv = rsqrtf(red[0] * (1.0f / D_MODEL) + 1e-5f);
    for (int c = threadIdx.x; c < D_MODEL; c += 256)
        xn[(size_t)row * D_MODEL + c] = (bf16)((xr[c] - mu) * inv * g[c] + b[c]);
}

/* depthwise causal conv (K=4) + SiLU; dir=1 runs time-reversed so the
 * outer flip of the backward branch cancels.
 * grid = (D_INNER/256, LSEQ, BATCH): no per-thread div/mod. */
__global__ void k_conv_silu(const float* __restrict__ xz,
                            const float* __restrict__ cw,
                            const float* __restrict__ cb,
                            bf16* __restrict__ u, int dir)
{
    const int d = blockIdx.x * blockDim.x + threadIdx.x;
    const int t = blockIdx.y;
    const int b = blockIdx.z;
    float acc = cb[d];
#pragma unroll
    for (int k = 0; k < KCONV; ++k) {
        const int tt = (dir == 0) ? (t - (KCONV - 1) + k) : (t + (KCONV - 1) - k);
        if (tt >= 0 && tt < LSEQ)
            acc += cw[d * KCONV + k] *
                   xz[((long)b * LSEQ + tt) * (2L * D_INNER) + d];
    }
    const float sg = 1.0f / (1.0f + __expf(-acc));
    u[((long)b * LSEQ + t) * D_INNER + d] = (bf16)(acc * sg);
}

/* selective scan: one thread per (batch, channel), h[16] in VGPRs.
 * Fused: softplus(delta), state update, C-contraction, D-skip, SiLU(z)
 * gate, bf16 emit for the w_out GEMM. dir=1 iterates t backwards. */
__global__ __launch_bounds__(256)
void k_scan(const bf16* __restrict__ u, const float* __restrict__ dpre,
            const float* __restrict__ dt_b, const float* __restrict__ dbl,
            const float* __restrict__ xz, const float* __restrict__ A_log,
            const float* __restrict__ Dskip, bf16* __restrict__ y, int dir)
{
    const int gid = blockIdx.x * 256 + threadIdx.x;
    const int b = gid / D_INNER;
    const int d = gid % D_INNER;

    float A[NSTATE], h[NSTATE];
#pragma unroll
    for (int n = 0; n < NSTATE; ++n) {
        A[n] = -__expf(A_log[(long)d * NSTATE + n]);
        h[n] = 0.0f;
    }
    const float dk  = Dskip[d];
    const float dtb = dt_b[d];

    for (int s = 0; s < LSEQ; ++s) {
        const int  t   = (dir == 0) ? s : (LSEQ - 1 - s);
        const long row = (long)b * LSEQ + t;

        float dl = dpre[row * D_INNER + d] + dtb;
        dl = (dl > 20.0f) ? dl : log1pf(__expf(dl));
        const float uv = (float)u[row * D_INNER + d];
        const float z  = xz[row * (2L * D_INNER) + D_INNER + d];

        const float4* Bp = (const float4*)(dbl + row * DBL_PAD + RANK);
        const float4* Cp = (const float4*)(dbl + row * DBL_PAD + RANK + NSTATE);
        float4 Bv[4] = {Bp[0], Bp[1], Bp[2], Bp[3]};
        float4 Cv[4] = {Cp[0], Cp[1], Cp[2], Cp[3]};
        const float* Bf = (const float*)Bv;
        const float* Cf = (const float*)Cv;

        const float du = dl * uv;
        float yv = 0.0f;
#pragma unroll
        for (int n = 0; n < NSTATE; ++n) {
            h[n] = h[n] * __expf(dl * A[n]) + du * Bf[n];
            yv += h[n] * Cf[n];
        }
        const float zg = z / (1.0f + __expf(-z));
        y[row * D_INNER + d] = (bf16)((yv + uv * dk) * zg);
    }
}

/* ------------------------------ driver ---------------------------- */
extern "C" void kernel_launch(void* const* d_in, const int* in_sizes, int n_in,
                              void* d_out, int out_size, void* d_ws, size_t ws_size,
                              hipStream_t stream)
{
    const float* x = (const float*)d_in[0];
    struct DirW { const float *w_in, *cw, *cb, *xp, *dtw, *dtb, *Alog, *Dk, *wout; };
    DirW dw[2];
    for (int t = 0; t < 2; ++t) {
        const int o = 1 + t * 9;
        dw[t].w_in = (const float*)d_in[o + 0];
        dw[t].cw   = (const float*)d_in[o + 1];
        dw[t].cb   = (const float*)d_in[o + 2];
        dw[t].xp   = (const float*)d_in[o + 3];
        dw[t].dtw  = (const float*)d_in[o + 4];
        dw[t].dtb  = (const float*)d_in[o + 5];
        dw[t].Alog = (const float*)d_in[o + 6];
        dw[t].Dk   = (const float*)d_in[o + 7];
        dw[t].wout = (const float*)d_in[o + 8];
    }
    const float* proj_w = (const float*)d_in[19];
    const float* proj_b = (const float*)d_in[20];
    const float* ln_g   = (const float*)d_in[21];
    const float* ln_b   = (const float*)d_in[22];

    char* wsp = (char*)d_ws;
    auto alloc = [&](size_t bytes) -> char* {
        char* p = wsp;
        wsp += (bytes + 255) & ~(size_t)255;
        return p;
    };
    bf16*  xn    = (bf16*)alloc((size_t)MROWS * D_MODEL * 2);
    bf16*  wb_in = (bf16*)alloc((size_t)2 * D_INNER * D_MODEL * 2);
    bf16*  wb_xp = (bf16*)alloc((size_t)DBL_PAD * D_INNER * 2);
    bf16*  wb_dt = (bf16*)alloc((size_t)D_INNER * RANK * 2);
    bf16*  wb_out= (bf16*)alloc((size_t)D_MODEL * D_INNER * 2);
    bf16*  wb_pj = (bf16*)alloc((size_t)D_MODEL * 2 * D_MODEL * 2);
    float* xz    = (float*)alloc((size_t)MROWS * 2 * D_INNER * 4);
    bf16*  ubuf  = (bf16*)alloc((size_t)MROWS * D_INNER * 2);
    float* dbl   = (float*)alloc((size_t)MROWS * DBL_PAD * 4);
    bf16*  dt16  = (bf16*)alloc((size_t)MROWS * RANK * 2);
    float* dpre  = (float*)alloc((size_t)MROWS * D_INNER * 4);
    bf16*  ybf   = (bf16*)alloc((size_t)MROWS * D_INNER * 2);
    bf16*  mixed = (bf16*)alloc((size_t)MROWS * 2 * D_MODEL * 2);

    const dim3 blk(256);
    auto g1 = [](long n) { return dim3((unsigned)((n + 255) / 256)); };

    k_ln<<<MROWS, blk, 0, stream>>>(x, ln_g, ln_b, xn);

    const long n_pj = (long)D_MODEL * 2 * D_MODEL;
    k_cvt<<<g1(n_pj), blk, 0, stream>>>(proj_w, wb_pj, n_pj);

    for (int dir = 0; dir < 2; ++dir) {
        const DirW& P = dw[dir];

        /* xz = xn @ w_in.T : 4096 x 4096, K=1024 */
        k_cvt<<<g1((long)2 * D_INNER * D_MODEL), blk, 0, stream>>>(
            P.w_in, wb_in, (long)2 * D_INNER * D_MODEL);
        gemm_wmma<<<dim3(2 * D_INNER / TN, MROWS / TM), blk, 0, stream>>>(
            xn, D_MODEL, wb_in, D_MODEL, xz, nullptr, 2 * D_INNER, 0,
            D_MODEL, nullptr, nullptr);

        /* u = silu(depthwise_conv(u_part)) */
        k_conv_silu<<<dim3(D_INNER / 256, LSEQ, BATCH), blk, 0, stream>>>(
            xz, P.cw, P.cb, ubuf, dir);

        /* dbl = u @ x_proj.T : N padded 96 -> 128, K=2048 */
        k_cvt_pad<<<g1((long)DBL_PAD * D_INNER), blk, 0, stream>>>(
            P.xp, wb_xp, RANK + 2 * NSTATE, D_INNER, (long)DBL_PAD * D_INNER);
        gemm_wmma<<<dim3(DBL_PAD / TN, MROWS / TM), blk, 0, stream>>>(
            ubuf, D_INNER, wb_xp, D_INNER, dbl, nullptr, DBL_PAD, 0,
            D_INNER, nullptr, nullptr);

        /* dpre = dt @ dt_w.T : 4096 x 2048, K=64 */
        k_slice_bf16<<<g1((long)MROWS * RANK), blk, 0, stream>>>(
            dbl, DBL_PAD, dt16, RANK, RANK, MROWS);
        k_cvt<<<g1((long)D_INNER * RANK), blk, 0, stream>>>(
            P.dtw, wb_dt, (long)D_INNER * RANK);
        gemm_wmma<<<dim3(D_INNER / TN, MROWS / TM), blk, 0, stream>>>(
            dt16, RANK, wb_dt, RANK, dpre, nullptr, D_INNER, 0,
            RANK, nullptr, nullptr);

        /* selective scan + gate, bf16 out */
        k_scan<<<(BATCH * D_INNER) / 256, blk, 0, stream>>>(
            ubuf, dpre, P.dtb, dbl, xz, P.Alog, P.Dk, ybf, dir);

        /* mixed[:, dir*1024 : ] = y @ w_out.T : 4096 x 1024, K=2048 */
        k_cvt<<<g1((long)D_MODEL * D_INNER), blk, 0, stream>>>(
            P.wout, wb_out, (long)D_MODEL * D_INNER);
        gemm_wmma<<<dim3(D_MODEL / TN, MROWS / TM), blk, 0, stream>>>(
            ybf, D_INNER, wb_out, D_INNER, nullptr, mixed, 2 * D_MODEL,
            dir * D_MODEL, D_INNER, nullptr, nullptr);
    }

    /* out = x + mixed @ proj_w.T + proj_b : 4096 x 1024, K=2048 */
    gemm_wmma<<<dim3(D_MODEL / TN, MROWS / TM), blk, 0, stream>>>(
        mixed, 2 * D_MODEL, wb_pj, 2 * D_MODEL, (float*)d_out, nullptr,
        D_MODEL, 0, 2 * D_MODEL, proj_b, x);

    (void)in_sizes; (void)n_in; (void)out_size; (void)ws_size;
}
